// CausalSelfAttention_38809324486673
// MI455X (gfx1250) — matrix operand order repaired and checked by reading, not compiled, not run on hardware
//
#include <hip/hip_runtime.h>
#include <hip/hip_bf16.h>

// ---------------------------------------------------------------------------
// CDNA5 (gfx1250) bf16 WMMA attention:  y = proj( softmax(QK^T/sqrt(hd)) V )
//   D=1024, H=16, hd=64, T=4096, B=1   (non-causal, per reference)
// Data movement: GLOBAL_LOAD_ASYNC_TO_LDS_B128 (GEMM staging, ASYNCcnt) and
// TDM TENSOR_LOAD_TO_LDS (flash K-tile, TENSORcnt) + global_prefetch.
// ---------------------------------------------------------------------------

typedef __bf16 bf16;
typedef __attribute__((ext_vector_type(16))) __bf16 v16bf;
typedef __attribute__((ext_vector_type(8)))  float  v8f;
typedef __attribute__((ext_vector_type(4)))  unsigned int u32x4;
typedef __attribute__((ext_vector_type(8)))  int i32x8;
typedef __attribute__((ext_vector_type(4)))  int i32x4;

#define Tlen 4096
#define Dmod 1024
#define Hn   16
#define HD   64

// ---------------------------------------------------------------------------
// Fragment loaders following the CDNA5 WMMA VGPR layouts (ISA 7.12.2).
// ---------------------------------------------------------------------------
__device__ __forceinline__ v16bf load_frag_a(const bf16* base, int ld, int lane) {
    union { v16bf v; uint4 q[2]; } u;
    const bf16* p = base + (lane & 15) * ld + ((lane < 16) ? 0 : 8);
    u.q[0] = *reinterpret_cast<const uint4*>(p);
    u.q[1] = *reinterpret_cast<const uint4*>(p + 16);
    return u.v;
}

__device__ __forceinline__ v16bf load_frag_b(const bf16* base, int ld, int lane) {
    union { v16bf v; uint4 q[2]; } u;
    const bf16* p = base + (lane & 15) * ld + ((lane < 16) ? 0 : 8);
    u.q[0] = *reinterpret_cast<const uint4*>(p);
    u.q[1] = *reinterpret_cast<const uint4*>(p + 16);
    return u.v;
}

__device__ __forceinline__ v8f wmma_bf16(v16bf a, v16bf b, v8f c) {
    return __builtin_amdgcn_wmma_f32_16x16x32_bf16(
        false, a, false, b, (short)0, c, false, false);
}

// ---------------------------------------------------------------------------
// CDNA5 async copy: GLOBAL_LOAD_ASYNC_TO_LDS_B128 (per-lane 16B, ASYNCcnt)
// Generic LDS pointer low 32 bits are the LDS byte address (aperture map).
// ---------------------------------------------------------------------------
__device__ __forceinline__ void async_copy_b128(void* lds_dst, const void* gsrc) {
    unsigned lds_addr = (unsigned)(size_t)lds_dst;
    unsigned long long ga = (unsigned long long)(size_t)gsrc;
    asm volatile("global_load_async_to_lds_b128 %0, %1, off"
                 :: "v"(lds_addr), "v"(ga) : "memory");
}

__device__ __forceinline__ void wait_async0() {
    asm volatile("s_wait_asynccnt 0x0" ::: "memory");
}

// ---------------------------------------------------------------------------
// f32 -> bf16 bulk conversion (4 elements / thread, 8B packed store)
// ---------------------------------------------------------------------------
__global__ __launch_bounds__(256) void f32_to_bf16_kernel(
    const float* __restrict__ src, bf16* __restrict__ dst, int n)
{
    int i = (blockIdx.x * 256 + threadIdx.x) * 4;
    if (i < n) {
        float4 f = *reinterpret_cast<const float4*>(src + i);
        union { bf16 h[4]; uint2 u; } o;
        o.h[0] = (bf16)f.x; o.h[1] = (bf16)f.y;
        o.h[2] = (bf16)f.z; o.h[3] = (bf16)f.w;
        *reinterpret_cast<uint2*>(dst + i) = o.u;
    }
}

// ---------------------------------------------------------------------------
// bf16 GEMM:  C[m,n] = sum_k A[m,k] * B[n,k]   (B is [N,K] row-major weight)
// Block tile 128(M) x 64(N), K-slab 64, 8 waves (4x2 of 32x32 wave tiles).
// LDS staging via GLOBAL_LOAD_ASYNC_TO_LDS_B128.
// mode 0: f32 C row-major -> Cout ; mode 1: QKV scatter (Q scaled 0.125)
// ---------------------------------------------------------------------------
__global__ __launch_bounds__(256) void gemm_bf16_kernel(
    const bf16* __restrict__ A, const bf16* __restrict__ B,
    int N, int Kdim, int mode,
    float* __restrict__ Cout,
    bf16* __restrict__ Qb, bf16* __restrict__ Kb, bf16* __restrict__ Vb)
{
    constexpr int BM = 128, BN = 64, BK = 64, LDK = BK + 8;   // 144B row pitch
    __shared__ alignas(16) bf16 As[BM * LDK];
    __shared__ alignas(16) bf16 Bs[BN * LDK];

    const int tid  = threadIdx.x;
    const int lane = tid & 31;
    const int wave = tid >> 5;
    const int wm   = wave & 3;
    const int wn   = wave >> 2;
    const int m0   = blockIdx.y * BM;
    const int n0   = blockIdx.x * BN;

    v8f acc[2][2] = {};

    for (int kb = 0; kb < Kdim; kb += BK) {
        // stage A tile: 128x64 bf16 = 1024 x 16B async transfers, 4/thread
#pragma unroll
        for (int v = 0; v < 4; ++v) {
            int slot = tid + v * 256;
            int r = slot >> 3, c = (slot & 7) * 8;
            async_copy_b128(&As[r * LDK + c], &A[(m0 + r) * Kdim + kb + c]);
        }
        // stage B tile: 64x64 bf16 = 512 x 16B async transfers, 2/thread
#pragma unroll
        for (int v = 0; v < 2; ++v) {
            int slot = tid + v * 256;
            int r = slot >> 3, c = (slot & 7) * 8;
            async_copy_b128(&Bs[r * LDK + c], &B[(n0 + r) * Kdim + kb + c]);
        }
        wait_async0();
        __syncthreads();

#pragma unroll
        for (int ks = 0; ks < BK; ks += 32) {
            v16bf a0 = load_frag_a(&As[(wm * 32 +  0) * LDK + ks], LDK, lane);
            v16bf a1 = load_frag_a(&As[(wm * 32 + 16) * LDK + ks], LDK, lane);
            v16bf b0 = load_frag_b(&Bs[(wn * 32 +  0) * LDK + ks], LDK, lane);
            v16bf b1 = load_frag_b(&Bs[(wn * 32 + 16) * LDK + ks], LDK, lane);
            acc[0][0] = wmma_bf16(a0, b0, acc[0][0]);
            acc[0][1] = wmma_bf16(a0, b1, acc[0][1]);
            acc[1][0] = wmma_bf16(a1, b0, acc[1][0]);
            acc[1][1] = wmma_bf16(a1, b1, acc[1][1]);
        }
        __syncthreads();
    }

    const int hi = (lane >> 4) * 8;
    const int nl = lane & 15;
#pragma unroll
    for (int mi = 0; mi < 2; ++mi)
#pragma unroll
        for (int ni = 0; ni < 2; ++ni)
#pragma unroll
            for (int r = 0; r < 8; ++r) {
                int m = m0 + wm * 32 + mi * 16 + hi + r;
                int n = n0 + wn * 32 + ni * 16 + nl;
                float val = acc[mi][ni][r];
                if (mode == 0) {
                    Cout[m * N + n] = val;
                } else {
                    int sec = n >> 10;            // 0:Q 1:K 2:V
                    int c   = n & (Dmod - 1);
                    int h   = c >> 6;
                    int d   = c & (HD - 1);
                    long idx = ((long)h * Tlen + m) * HD + d;
                    if (sec == 0)      Qb[idx] = (bf16)(val * 0.125f); // 1/sqrt(64)
                    else if (sec == 1) Kb[idx] = (bf16)val;
                    else               Vb[idx] = (bf16)val;
                }
            }
}

// ---------------------------------------------------------------------------
// Flash attention (non-causal): one block per (64-query tile, head).
// 128 threads = 4 waves; each wave owns 16 query rows x hd=64.
// K tile staged by the Tensor Data Mover (TENSOR_LOAD_TO_LDS, 2-D D#,
// pad_enable producing the 72-element padded LDS pitch); V staged manually
// transposed. Per 64-key tile: 8 WMMA (QK^T) + 8 WMMA (PV).
// ---------------------------------------------------------------------------
__global__ __launch_bounds__(128) void flash_attn_kernel(
    const bf16* __restrict__ Qb, const bf16* __restrict__ Kb,
    const bf16* __restrict__ Vb, bf16* __restrict__ Ob)
{
    constexpr int BJ = 64, LD = HD + 8;                 // 72-elt (144B) pitch
    __shared__ alignas(16) bf16 Kt[BJ * LD];            // K tile [j][d] (padded)
    __shared__ alignas(16) bf16 Vt[HD * LD];            // V tile transposed [d][j]
    __shared__ alignas(16) bf16 Pt[4 * 16 * HD];        // per-wave P tiles [16][64]

    const int tid  = threadIdx.x;
    const int lane = tid & 31;
    const int wave = tid >> 5;
    const int h    = blockIdx.y;
    const int q0   = blockIdx.x * 64 + wave * 16;

    const bf16* Qh = Qb + (long)h * Tlen * HD;
    const bf16* Kh = Kb + (long)h * Tlen * HD;
    const bf16* Vh = Vb + (long)h * Tlen * HD;
    bf16* Pw = &Pt[wave * 16 * HD];

    v16bf qf[2];
#pragma unroll
    for (int ks = 0; ks < 2; ++ks)
        qf[ks] = load_frag_a(Qh + q0 * HD + ks * 32, HD, lane);

    v8f   o[4] = {};
    float mrow[8], lrow[8];
#pragma unroll
    for (int r = 0; r < 8; ++r) { mrow[r] = -3.0e38f; lrow[r] = 0.0f; }

    for (int jb = 0; jb < Tlen; jb += BJ) {
        // ---- TDM: stage 64x64 bf16 K tile into padded LDS (wave 0 only) ----
        if (wave == 0) {
            unsigned long long ga = (unsigned long long)(size_t)(Kh + (size_t)jb * HD);
            u32x4 g0;
            g0.x = 1u;                                   // count=1, user D#
            g0.y = (unsigned)(size_t)&Kt[0];             // lds_addr
            g0.z = (unsigned)(ga & 0xFFFFFFFFu);         // global_addr[31:0]
            g0.w = (unsigned)((ga >> 32) & 0x1FFFFFFu)   // global_addr[56:32]
                 | 0x80000000u;                          // type=2 ("image")
            i32x8 g1;
            g1[0] = (1 << 16)                            // data_size=2B
                  | (1 << 20)                            // pad_enable
                  | (4 << 22)                            // pad_interval: 32 DWORDs
                  | (3 << 25);                           // pad_amount:   4 DWORDs
            g1[1] = (HD & 0xFFFF) << 16;                 // tensor_dim0 = 64
            g1[2] = (Tlen & 0xFFFF) << 16;               // tensor_dim1 = 4096
            g1[3] = (HD & 0xFFFF) << 16;                 // tile_dim0 = 64
            g1[4] = BJ;                                  // tile_dim1 = 64
            g1[5] = HD;                                  // tensor_dim0_stride = 64
            g1[6] = 0;
            g1[7] = 0;
            i32x4 gz4 = {0, 0, 0, 0};
            i32x8 gz8 = {0, 0, 0, 0, 0, 0, 0, 0};
            __builtin_amdgcn_tensor_load_to_lds(g0, g1, gz4, gz4, gz8, 0);
        }

        // ---- stage V tile transposed: read 8 contiguous, scatter by d ----
#pragma unroll
        for (int v = 0; v < 4; ++v) {
            int slot = tid + v * 128;
            int j = slot >> 3, c = (slot & 7) * 8;
            union { uint4 q; bf16 e[8]; } u;
            u.q = *reinterpret_cast<const uint4*>(&Vh[(jb + j) * HD + c]);
#pragma unroll
            for (int i = 0; i < 8; ++i)
                Vt[(c + i) * LD + j] = u.e[i];
        }

        // prefetch next tiles into cache while this one computes
        if (jb + BJ < Tlen) {
            __builtin_prefetch(&Kh[(jb + BJ + (tid >> 1)) * HD], 0, 0);
            __builtin_prefetch(&Vh[(jb + BJ + (tid >> 1)) * HD], 0, 0);
        }

        if (wave == 0)
            __builtin_amdgcn_s_wait_tensorcnt(0);
        __syncthreads();

        // ---- S = Q K^T  (16 x 64 f32 per wave) ----
        v8f s[4] = {};
#pragma unroll
        for (int ks = 0; ks < 2; ++ks) {
#pragma unroll
            for (int nt = 0; nt < 4; ++nt) {
                v16bf bk = load_frag_b(&Kt[(nt * 16) * LD + ks * 32], LD, lane);
                s[nt] = wmma_bf16(qf[ks], bk, s[nt]);
            }
        }

        // ---- online softmax: per lane, 8 rows (r), 4 cols/row here ----
#pragma unroll
        for (int r = 0; r < 8; ++r) {
            float rm = fmaxf(fmaxf(s[0][r], s[1][r]), fmaxf(s[2][r], s[3][r]));
            rm = fmaxf(rm, __shfl_xor(rm, 1, 32));
            rm = fmaxf(rm, __shfl_xor(rm, 2, 32));
            rm = fmaxf(rm, __shfl_xor(rm, 4, 32));
            rm = fmaxf(rm, __shfl_xor(rm, 8, 32));
            float mnew  = fmaxf(mrow[r], rm);
            float alpha = __expf(mrow[r] - mnew);
            float rs = 0.0f;
#pragma unroll
            for (int nt = 0; nt < 4; ++nt) {
                float p = __expf(s[nt][r] - mnew);
                s[nt][r] = p;
                rs += p;
            }
            rs += __shfl_xor(rs, 1, 32);
            rs += __shfl_xor(rs, 2, 32);
            rs += __shfl_xor(rs, 4, 32);
            rs += __shfl_xor(rs, 8, 32);
            lrow[r] = lrow[r] * alpha + rs;
            mrow[r] = mnew;
#pragma unroll
            for (int nt = 0; nt < 4; ++nt)
                o[nt][r] *= alpha;
        }

        // ---- P (C layout) -> LDS -> A layout, private per wave ----
        const int hi = (lane >> 4) * 8;
        const int nl = lane & 15;
#pragma unroll
        for (int nt = 0; nt < 4; ++nt)
#pragma unroll
            for (int r = 0; r < 8; ++r)
                Pw[(hi + r) * HD + nt * 16 + nl] = (bf16)s[nt][r];

        // ---- O += P V ----
#pragma unroll
        for (int ks = 0; ks < 2; ++ks) {
            v16bf pa = load_frag_a(Pw + ks * 32, HD, lane);
#pragma unroll
            for (int nt = 0; nt < 4; ++nt) {
                v16bf bv = load_frag_b(&Vt[(nt * 16) * LD + ks * 32], LD, lane);
                o[nt] = wmma_bf16(pa, bv, o[nt]);
            }
        }
        __syncthreads();   // before next tile overwrites Kt/Vt
    }

    // ---- normalize, store O as bf16 into [T, D] at column h*64 ----
    const int hi = (lane >> 4) * 8;
    const int nl = lane & 15;
#pragma unroll
    for (int r = 0; r < 8; ++r) {
        float inv = 1.0f / lrow[r];
        int row = q0 + hi + r;
#pragma unroll
        for (int nt = 0; nt < 4; ++nt)
            Ob[row * Dmod + h * HD + nt * 16 + nl] = (bf16)(o[nt][r] * inv);
    }
}

// ---------------------------------------------------------------------------
// Launcher
// ---------------------------------------------------------------------------
extern "C" void kernel_launch(void* const* d_in, const int* in_sizes, int n_in,
                              void* d_out, int out_size, void* d_ws, size_t ws_size,
                              hipStream_t stream) {
    const float* x      = (const float*)d_in[0];   // [1,4096,1024]
    const float* w_attn = (const float*)d_in[1];   // [3072,1024]
    const float* w_proj = (const float*)d_in[2];   // [1024,1024]
    float* out = (float*)d_out;                    // [1,4096,1024]

    // workspace carve-up (bf16 buffers), total = 48 MB
    char* ws = (char*)d_ws;
    bf16* Xb  = (bf16*)ws;  ws += (size_t)Tlen * Dmod * 2;
    bf16* Wab = (bf16*)ws;  ws += (size_t)3 * Dmod * Dmod * 2;
    bf16* Wpb = (bf16*)ws;  ws += (size_t)Dmod * Dmod * 2;
    bf16* Qb  = (bf16*)ws;  ws += (size_t)Hn * Tlen * HD * 2;
    bf16* Kb  = (bf16*)ws;  ws += (size_t)Hn * Tlen * HD * 2;
    bf16* Vb  = (bf16*)ws;  ws += (size_t)Hn * Tlen * HD * 2;
    bf16* Ob  = (bf16*)ws;  ws += (size_t)Tlen * Dmod * 2;

    {
        int n1 = Tlen * Dmod;
        int n2 = 3 * Dmod * Dmod;
        int n3 = Dmod * Dmod;
        f32_to_bf16_kernel<<<dim3((n1 / 4 + 255) / 256), dim3(256), 0, stream>>>(x, Xb, n1);
        f32_to_bf16_kernel<<<dim3((n2 / 4 + 255) / 256), dim3(256), 0, stream>>>(w_attn, Wab, n2);
        f32_to_bf16_kernel<<<dim3((n3 / 4 + 255) / 256), dim3(256), 0, stream>>>(w_proj, Wpb, n3);
    }

    gemm_bf16_kernel<<<dim3(3 * Dmod / 64, Tlen / 128), dim3(256), 0, stream>>>(
        Xb, Wab, 3 * Dmod, Dmod, 1, nullptr, Qb, Kb, Vb);

    flash_attn_kernel<<<dim3(Tlen / 64, Hn), dim3(128), 0, stream>>>(Qb, Kb, Vb, Ob);

    gemm_bf16_kernel<<<dim3(Dmod / 64, Tlen / 128), dim3(256), 0, stream>>>(
        Ob, Wpb, Dmod, Dmod, 0, out, nullptr, nullptr, nullptr);

    (void)in_sizes; (void)n_in; (void)out_size; (void)ws_size;
}
